// PointNextPatchTokenizer_11407433138454
// MI455X (gfx1250) — compile-verified, matrix operand order checked by reading, not run on hardware
//
#include <hip/hip_runtime.h>
#include <hip/hip_bf16.h>
#include <cstdint>

// ---------------------------------------------------------------------------
// PointNeXt patch tokenizer for MI455X (gfx1250, wave32, WMMA).
// - All MLP GEMMs: v_wmma_f32_16x16x32_f16, A staged in LDS, weights
//   pre-swizzled into WMMA B-fragment layout (8 contiguous dwords/lane).
// - Fully static tiling: wave id forced scalar (readfirstlane), tile pairs
//   proven by static_assert -> no exec-masked WMMA regions.
// - Neighbor-feature gathers staged into LDS A-tiles with
//   global_load_async_to_lds_b128 (ASYNCcnt), overlapped with VALU PE build.
// - Parallel LayerNorm via __shfl_xor row reductions.
// - Ball query = wave32 ballot ordered compaction (== reference semantics).
// ---------------------------------------------------------------------------

#define B_      4
#define N_      4096
#define CIN_    6
#define STEM_   64
#define TOK_    128
#define NPATCH_ 64
#define GK_     32
#define LN_EPS_ 1e-5f

typedef __attribute__((ext_vector_type(16))) _Float16 v16h;
typedef __attribute__((ext_vector_type(8)))  float    v8f;

union F16x16 { v16h v; unsigned u[8]; _Float16 h[16]; };

// ---- helpers ---------------------------------------------------------------

__device__ __forceinline__ int wmma_kbase(int v, int hi) {
  // 16-bit A/B fragment K mapping (ISA 7.12.2): VGPR v holds K pairs
  return ((v < 4) ? (2 * v) : (16 + 2 * (v - 4))) + 8 * hi;
}

__device__ __forceinline__ float gelu_exact(float x) {
  return 0.5f * x * (1.0f + erff(x * 0.70710678118654752440f));
}

// order-preserving float <-> uint encoding for atomic max
__device__ __forceinline__ unsigned fenc(float x) {
  unsigned u = __float_as_uint(x);
  return (u & 0x80000000u) ? ~u : (u | 0x80000000u);
}
__device__ __forceinline__ float fdec(unsigned e) {
  return (e & 0x80000000u) ? __uint_as_float(e ^ 0x80000000u)
                           : __uint_as_float(~e);
}

// load a WMMA B fragment (pre-swizzled: 8 contiguous dwords per lane)
__device__ __forceinline__ v16h load_bfrag(const unsigned* __restrict__ Wf,
                                           int kt, int nt, int lane) {
  F16x16 u;
  const unsigned* p = Wf + ((size_t)(kt * 8 + nt) * 32 + lane) * 8;
#pragma unroll
  for (int v = 0; v < 8; ++v) u.u[v] = p[v];
  return u.v;
}

// load a WMMA A fragment from LDS (row-major, KP halves per row)
__device__ __forceinline__ v16h load_afrag(const _Float16* ldsA, int KP,
                                           int mbase, int kt, int lane) {
  int m  = mbase + (lane & 15);
  int hi = lane >> 4;
  F16x16 u;
#pragma unroll
  for (int v = 0; v < 8; ++v) {
    int k0 = kt * 32 + wmma_kbase(v, hi);
    u.h[2 * v]     = ldsA[m * KP + k0];
    u.h[2 * v + 1] = ldsA[m * KP + k0 + 1];
  }
  return u.v;
}

__device__ __forceinline__ void store_dtile(float* ldsY, int mbase, int nbase,
                                            int lane, v8f c) {
  int n  = nbase + (lane & 15);
  int hi = lane >> 4;
#pragma unroll
  for (int r = 0; r < 8; ++r) ldsY[(mbase + r + 8 * hi) * TOK_ + n] = c[r];
}

// Y(ROWS x 128) = A(ROWS x KP) * W via WMMA. Fully static tiling: each wave
// accumulates TWO independent (M,N) tiles interleaved in K (independent
// v_wmma streams hide the D->C chain). Wave id is forced scalar so all tile
// selection is SGPR math -> no exec-masked WMMA.
template <int KP, int ROWS, int NT>
__device__ __forceinline__ void wmma_layer(const _Float16* ldsA,
                                           const unsigned* __restrict__ Wf,
                                           float* ldsY, int tid) {
  constexpr int NW    = NT / 32;
  constexpr int KT    = KP / 32;
  constexpr int TILES = (ROWS / 16) * 8;
  static_assert(TILES % (2 * NW) == 0, "tiles must pair evenly across waves");
  int lane = tid & 31;
  int wave = __builtin_amdgcn_readfirstlane(tid >> 5);
#pragma unroll
  for (int it = 0; it < TILES / (2 * NW); ++it) {
    int t0 = wave + it * 2 * NW;
    int t1 = t0 + NW;
    int mi0 = t0 >> 3, ni0 = t0 & 7;
    int mi1 = t1 >> 3, ni1 = t1 & 7;
    v8f c0 = {}, c1 = {};
#pragma unroll
    for (int kt = 0; kt < KT; ++kt) {
      v16h a0 = load_afrag(ldsA, KP, mi0 * 16, kt, lane);
      v16h b0 = load_bfrag(Wf, kt, ni0, lane);
      v16h a1 = load_afrag(ldsA, KP, mi1 * 16, kt, lane);
      v16h b1 = load_bfrag(Wf, kt, ni1, lane);
      c0 = __builtin_amdgcn_wmma_f32_16x16x32_f16(false, a0, false, b0,
                                                  (short)0, c0, false, false);
      c1 = __builtin_amdgcn_wmma_f32_16x16x32_f16(false, a1, false, b1,
                                                  (short)0, c1, false, false);
    }
    store_dtile(ldsY, mi0 * 16, ni0 * 16, lane, c0);
    store_dtile(ldsY, mi1 * 16, ni1 * 16, lane, c1);
  }
}

// parallel bias + layernorm (+gelu). TPR threads per row (power of two,
// contiguous lanes), channels register-resident, __shfl_xor reduction.
// Requires nthreads == rows * TPR.
template <int TPR, bool ACT, bool TO_F16>
__device__ __forceinline__ void ln_rows_par(float* ldsY,
                                            const float* __restrict__ bb,
                                            const float* __restrict__ g,
                                            const float* __restrict__ be,
                                            _Float16* ldsA2, int tid) {
  constexpr int CPT = TOK_ / TPR;
  int row = tid / TPR, sub = tid % TPR;
  float* y = ldsY + row * TOK_ + sub * CPT;
  const int cb = sub * CPT;
  float v[CPT];
  float s = 0.f, ss = 0.f;
#pragma unroll
  for (int j = 0; j < CPT; ++j) {
    float t = y[j] + bb[cb + j];
    v[j] = t; s += t; ss += t * t;
  }
#pragma unroll
  for (int m = TPR >> 1; m > 0; m >>= 1) {
    s  += __shfl_xor(s, m, TPR);
    ss += __shfl_xor(ss, m, TPR);
  }
  float mu  = s * (1.f / TOK_);
  float var = ss * (1.f / TOK_) - mu * mu;
  float inv = rsqrtf(var + LN_EPS_);
#pragma unroll
  for (int j = 0; j < CPT; ++j) {
    float t = (v[j] - mu) * inv * g[cb + j] + be[cb + j];
    if (ACT) t = gelu_exact(t);
    if (TO_F16) ldsA2[row * TOK_ + cb + j] = (_Float16)t;
    else        y[j] = t;
  }
}

// wave32 ordered ball query over xyz (stride CIN_=6): first K in-radius
// indices in ascending order, padded with the first hit. Run by one wave.
// Counter kept scalar (ballot is uniform) so the scan is a scalar loop.
__device__ __forceinline__ void ball_query_wave(const float* __restrict__ pcb,
                                                float cx, float cy, float cz,
                                                float r2, int K, int* nbr,
                                                int lane) {
  int cnt = 0, first = 0;
  bool haveFirst = false;
  for (int base = 0; base < N_ && cnt < K; base += 32) {
    if (base + 128 < N_)
      __builtin_prefetch(pcb + (size_t)(base + 128) * CIN_, 0, 1);
    int j = base + lane;
    float dx = pcb[j * CIN_ + 0] - cx;
    float dy = pcb[j * CIN_ + 1] - cy;
    float dz = pcb[j * CIN_ + 2] - cz;
    bool in = (dx * dx + dy * dy + dz * dz) <= r2;
    unsigned mask = (unsigned)__ballot(in);
    if (mask) {
      if (!haveFirst) { first = base + __builtin_ctz(mask); haveFirst = true; }
      int rank = __builtin_popcount(mask & ((1u << lane) - 1u));
      if (in && (cnt + rank) < K) nbr[cnt + rank] = j;
      cnt = __builtin_amdgcn_readfirstlane(
          min(cnt + __builtin_popcount(mask), K));
    }
  }
  for (int i = cnt + lane; i < K; i += 32) nbr[i] = first;
}

// async 16B chunk DMA: global (pfh row) -> LDS A tile, ASYNCcnt tracked
__device__ __forceinline__ void async_chunk(unsigned ldsoff, unsigned voff,
                                            unsigned long long sbase) {
  asm volatile("global_load_async_to_lds_b128 %0, %1, %2"
               :: "v"(ldsoff), "v"(voff), "s"(sbase)
               : "memory");
}
__device__ __forceinline__ void async_wait0() {
  asm volatile("s_wait_asynccnt 0x0" ::: "memory");
}

// ---- kernel: weight -> WMMA B-fragment layout, f16, K padded ---------------

__global__ void k_prep_wfrag(const float* __restrict__ W, int Kact,
                             unsigned* __restrict__ dst) {
  int kt = blockIdx.x, nt = blockIdx.y, lane = threadIdx.x;
  int n  = nt * 16 + (lane & 15);
  int hi = lane >> 4;
  unsigned out[8];
#pragma unroll
  for (int v = 0; v < 8; ++v) {
    int k0 = kt * 32 + wmma_kbase(v, hi);
    float f0 = (k0 < Kact)     ? W[(size_t)k0 * TOK_ + n]       : 0.f;
    float f1 = (k0 + 1 < Kact) ? W[(size_t)(k0 + 1) * TOK_ + n] : 0.f;
    unsigned short h0 = __builtin_bit_cast(unsigned short, (_Float16)f0);
    unsigned short h1 = __builtin_bit_cast(unsigned short, (_Float16)f1);
    out[v] = ((unsigned)h1 << 16) | (unsigned)h0;
  }
  unsigned* p = dst + ((size_t)(kt * 8 + nt) * 32 + lane) * 8;
#pragma unroll
  for (int v = 0; v < 8; ++v) p[v] = out[v];
}

__global__ void k_init_gmax(unsigned* __restrict__ gmax) {
  gmax[threadIdx.x] = fenc(-3.4e38f);
}

// ---- kernel: stem MLP (6->64 gelu, 64->64 gelu), writes pfh (f16) ----------

__global__ __launch_bounds__(64)
void k_stem(const float* __restrict__ pc,
            const float* __restrict__ W1, const float* __restrict__ b1,
            const float* __restrict__ g1, const float* __restrict__ be1,
            const float* __restrict__ W2, const float* __restrict__ b2,
            const float* __restrict__ g2, const float* __restrict__ be2,
            _Float16* __restrict__ pfh) {
  __shared__ float sW1[CIN_ * STEM_];
  __shared__ float sW2[STEM_ * STEM_];
  __shared__ float sb1[STEM_], sg1[STEM_], sbe1[STEM_];
  __shared__ float sb2[STEM_], sg2[STEM_], sbe2[STEM_];
  __shared__ float hbuf[64 * STEM_];
  __shared__ float obuf[64 * STEM_];
  int tid = threadIdx.x;
  for (int i = tid; i < CIN_ * STEM_; i += 64) sW1[i] = W1[i];
  for (int i = tid; i < STEM_ * STEM_; i += 64) sW2[i] = W2[i];
  sb1[tid] = b1[tid]; sg1[tid] = g1[tid]; sbe1[tid] = be1[tid];
  sb2[tid] = b2[tid]; sg2[tid] = g2[tid]; sbe2[tid] = be2[tid];
  __syncthreads();
  size_t p = (size_t)blockIdx.x * 64 + tid;  // global point id (0..B*N-1)
  float f[CIN_];
#pragma unroll
  for (int i = 0; i < CIN_; ++i) f[i] = pc[p * CIN_ + i];
  float s = 0.f, ss = 0.f;
  for (int j = 0; j < STEM_; ++j) {
    float a = sb1[j];
#pragma unroll
    for (int i = 0; i < CIN_; ++i) a += f[i] * sW1[i * STEM_ + j];
    hbuf[tid * STEM_ + j] = a; s += a; ss += a * a;
  }
  float mu = s / STEM_, var = ss / STEM_ - mu * mu;
  float inv = rsqrtf(var + LN_EPS_);
  for (int j = 0; j < STEM_; ++j) {
    float v = gelu_exact((hbuf[tid * STEM_ + j] - mu) * inv * sg1[j] + sbe1[j]);
    hbuf[tid * STEM_ + j] = v;
  }
  s = 0.f; ss = 0.f;
  for (int j = 0; j < STEM_; ++j) {
    float a = sb2[j];
    for (int i = 0; i < STEM_; ++i)
      a += hbuf[tid * STEM_ + i] * sW2[i * STEM_ + j];
    obuf[tid * STEM_ + j] = a; s += a; ss += a * a;
  }
  mu = s / STEM_; var = ss / STEM_ - mu * mu; inv = rsqrtf(var + LN_EPS_);
  for (int j = 0; j < STEM_; ++j) {
    float v = gelu_exact((obuf[tid * STEM_ + j] - mu) * inv * sg2[j] + sbe2[j]);
    pfh[p * STEM_ + j] = (_Float16)v;
  }
}

// ---- kernel: farthest point sampling (1 block per batch) -------------------

__global__ __launch_bounds__(256)
void k_fps(const float* __restrict__ pc, float* __restrict__ centers,
           int* __restrict__ cidx, float* __restrict__ out_centers) {
  __shared__ float dist[N_];
  __shared__ float rmax[256];
  __shared__ int   rarg[256];
  __shared__ int   s_far;
  int b = blockIdx.x, tid = threadIdx.x;
  const float* pcb = pc + (size_t)b * N_ * CIN_;
  for (int i = tid; i < N_; i += 256) dist[i] = 1e10f;
  if (tid == 0) s_far = 0;
  __syncthreads();
  for (int it = 0; it < NPATCH_; ++it) {
    int far = s_far;
    float cx = pcb[far * CIN_ + 0], cy = pcb[far * CIN_ + 1],
          cz = pcb[far * CIN_ + 2];
    if (tid == 0) {
      int o = b * NPATCH_ + it;
      cidx[o] = far;
      centers[o * 3 + 0] = cx; centers[o * 3 + 1] = cy; centers[o * 3 + 2] = cz;
      out_centers[o * 3 + 0] = cx; out_centers[o * 3 + 1] = cy;
      out_centers[o * 3 + 2] = cz;
    }
    float lm = -1.f; int la = 0;
    for (int i = tid; i < N_; i += 256) {
      float dx = pcb[i * CIN_ + 0] - cx, dy = pcb[i * CIN_ + 1] - cy,
            dz = pcb[i * CIN_ + 2] - cz;
      float d  = dx * dx + dy * dy + dz * dz;
      float nd = fminf(dist[i], d);
      dist[i]  = nd;
      if (nd > lm) { lm = nd; la = i; }
    }
    rmax[tid] = lm; rarg[tid] = la;
    __syncthreads();
    for (int st = 128; st > 0; st >>= 1) {
      if (tid < st) {
        bool take = (rmax[tid + st] > rmax[tid]) ||
                    (rmax[tid + st] == rmax[tid] && rarg[tid + st] < rarg[tid]);
        if (take) { rmax[tid] = rmax[tid + st]; rarg[tid] = rarg[tid + st]; }
      }
      __syncthreads();
    }
    if (tid == 0) s_far = rarg[0];
    __syncthreads();
  }
}

// ---- kernel: per-center scale branch (ball query + MLP2 + max) -------------

template <int K>
__global__ __launch_bounds__(128)
void k_scale(const float* __restrict__ pc, const _Float16* __restrict__ pfh,
             const float* __restrict__ centers,
             const unsigned* __restrict__ Wf1, const float* __restrict__ b1,
             const float* __restrict__ g1, const float* __restrict__ be1,
             const unsigned* __restrict__ Wf2, const float* __restrict__ b2,
             const float* __restrict__ g2, const float* __restrict__ be2,
             float r2, float* __restrict__ sfeat) {
  constexpr int ROWS = (K <= 16) ? 16 : 32;
  constexpr int TPR  = 128 / ROWS;
  __shared__ int nbr[32];
  __shared__ alignas(16) _Float16 A[32 * 96];
  __shared__ alignas(16) float    Y[32 * TOK_];
  __shared__ alignas(16) _Float16 A2[32 * TOK_];
  int tid = threadIdx.x;
  int bc = blockIdx.x;              // b*64 + center
  int b  = bc / NPATCH_;
  const float* pcb = pc + (size_t)b * N_ * CIN_;
  float cx = centers[bc * 3 + 0], cy = centers[bc * 3 + 1],
        cz = centers[bc * 3 + 2];
  if (tid < 32) ball_query_wave(pcb, cx, cy, cz, r2, K, nbr, tid);
  __syncthreads();
  // async DMA of neighbor features (dims 0..63) into A rows
  {
    int wv = __builtin_amdgcn_readfirstlane(tid >> 5);  // 4 waves
    int lane = tid & 31;
    if (lane < ROWS) {
      int nj = nbr[lane];
      unsigned long long sbase =
          (unsigned long long)(uintptr_t)(pfh + (size_t)b * N_ * STEM_);
#pragma unroll
      for (int c = 0; c < 2; ++c) {
        int ch = wv + c * 4;                 // 8 chunks of 16B per row
        unsigned ldsoff = (unsigned)(uintptr_t)&A[lane * 96 + ch * 8];
        unsigned voff   = (unsigned)(nj * STEM_ + ch * 8) * 2u;
        async_chunk(ldsoff, voff, sbase);
      }
    }
  }
  // build dims 64..95 (rel, rel_pe, pad) with VALU
  {
    int row = tid >> 2, seg = tid & 3;   // 4 threads/row * 8 dims
    if (row < ROWS) {
      int nj = nbr[row];
      float rx = pcb[nj * CIN_ + 0] - cx;
      float ry = pcb[nj * CIN_ + 1] - cy;
      float rz = pcb[nj * CIN_ + 2] - cz;
      for (int d = 64 + seg * 8; d < 64 + seg * 8 + 8; ++d) {
        _Float16 v;
        if (d < 67) {
          float rv = (d == 64) ? rx : ((d == 65) ? ry : rz);
          v = (_Float16)rv;
        } else if (d < 91) {
          int q = d - 67, c = q >> 3, r = q & 7, i = r & 3;
          float fr = (i == 0) ? 1.f : ((i == 1) ? 0.046415888f
                       : ((i == 2) ? 0.0021544347f : 1e-4f));
          float rv = (c == 0) ? rx : ((c == 1) ? ry : rz);
          float a  = rv * fr;
          v = (_Float16)((r < 4) ? __sinf(a) : __cosf(a));
        } else v = (_Float16)0.f;
        A[row * 96 + d] = v;
      }
    }
  }
  async_wait0();
  __syncthreads();
  wmma_layer<96, ROWS, 128>(A, Wf1, Y, tid);
  __syncthreads();
  ln_rows_par<TPR, true, true>(Y, b1, g1, be1, A2, tid);
  __syncthreads();
  wmma_layer<128, ROWS, 128>(A2, Wf2, Y, tid);
  __syncthreads();
  ln_rows_par<TPR, false, false>(Y, b2, g2, be2, nullptr, tid);
  __syncthreads();
  if (tid < TOK_) {
    float m = -3.4e38f;
#pragma unroll
    for (int r = 0; r < K; ++r) m = fmaxf(m, Y[r * TOK_ + tid]);
    sfeat[(size_t)bc * TOK_ + tid] = m;
  }
}

// ---- kernel: fused scene branch (query + MLP2 + max + global atomic max) ---

__global__ __launch_bounds__(256)
void k_scene(const float* __restrict__ pc, const _Float16* __restrict__ pfh,
             const unsigned* __restrict__ Wf1, const float* __restrict__ b1,
             const float* __restrict__ g1, const float* __restrict__ be1,
             const unsigned* __restrict__ Wf2, const float* __restrict__ b2,
             const float* __restrict__ g2, const float* __restrict__ be2,
             unsigned* __restrict__ gmax) {
  __shared__ int nbr[GK_];
  __shared__ float absbuf[96];
  __shared__ alignas(16) _Float16 A[32 * 192];
  __shared__ alignas(16) float    Y[32 * TOK_];
  __shared__ alignas(16) _Float16 A2[32 * TOK_];
  int tid = threadIdx.x;
  int gp = blockIdx.x;                 // 0..B*N-1
  int b = gp / N_, p = gp % N_;
  const float* pcb = pc + (size_t)b * N_ * CIN_;
  float cx = pcb[p * CIN_ + 0], cy = pcb[p * CIN_ + 1], cz = pcb[p * CIN_ + 2];
  if (tid < 32)
    ball_query_wave(pcb, cx, cy, cz, 0.16f * 0.16f, GK_, nbr, tid);
  if (tid >= 96 && tid < 192) {        // off wave0: wave0 runs the query
    int q = tid - 96;
    int c = q >> 5, r = q & 31, i = r & 15;
    float fr = __expf(-(float)i * (9.2103403720f / 15.f));
    float coord = (c == 0) ? cx : ((c == 1) ? cy : cz);
    float a = coord * fr;
    absbuf[q] = (r < 16) ? __sinf(a) : __cosf(a);
  }
  __syncthreads();
  // Async stage of neighbor features into the A tile: wave w copies 16B
  // chunk w of every row (lane == row). ASYNCcnt-tracked LDS DMA.
  {
    int wv = __builtin_amdgcn_readfirstlane(tid >> 5);  // 8 waves
    int lane = tid & 31;
    int nj = nbr[lane];
    unsigned ldsoff = (unsigned)(uintptr_t)&A[lane * 192 + wv * 8];
    unsigned voff   = (unsigned)(nj * STEM_ + wv * 8) * 2u;
    unsigned long long sbase =
        (unsigned long long)(uintptr_t)(pfh + (size_t)b * N_ * STEM_);
    async_chunk(ldsoff, voff, sbase);
  }
  // Build dims 64..191 (rel, rel_pe, abs_pe, pad) with VALU, overlapped
  // with the async copies of dims 0..63.
  {
    int row = tid >> 3, seg = tid & 7;  // 8 threads/row * 16 dims = 128
    int nj = nbr[row];
    float rx = pcb[nj * CIN_ + 0] - cx;
    float ry = pcb[nj * CIN_ + 1] - cy;
    float rz = pcb[nj * CIN_ + 2] - cz;
    for (int d = 64 + seg * 16; d < 64 + seg * 16 + 16; ++d) {
      _Float16 v;
      if (d < 67) {
        float rv = (d == 64) ? rx : ((d == 65) ? ry : rz);
        v = (_Float16)rv;
      } else if (d < 91) {
        int q = d - 67, c = q >> 3, r = q & 7, i = r & 3;
        float fr = (i == 0) ? 1.f : ((i == 1) ? 0.046415888f
                     : ((i == 2) ? 0.0021544347f : 1e-4f));
        float rv = (c == 0) ? rx : ((c == 1) ? ry : rz);
        float a  = rv * fr;
        v = (_Float16)((r < 4) ? __sinf(a) : __cosf(a));
      } else if (d < 187) v = (_Float16)absbuf[d - 91];
      else v = (_Float16)0.f;
      A[row * 192 + d] = v;
    }
  }
  async_wait0();
  __syncthreads();
  wmma_layer<192, 32, 256>(A, Wf1, Y, tid);
  __syncthreads();
  ln_rows_par<8, true, true>(Y, b1, g1, be1, A2, tid);
  __syncthreads();
  wmma_layer<128, 32, 256>(A2, Wf2, Y, tid);
  __syncthreads();
  ln_rows_par<8, false, false>(Y, b2, g2, be2, nullptr, tid);
  __syncthreads();
  if (tid < TOK_) {
    float m = -3.4e38f;
#pragma unroll
    for (int r = 0; r < GK_; ++r) m = fmaxf(m, Y[r * TOK_ + tid]);
    atomicMax(&gmax[b * TOK_ + tid], fenc(m));
  }
}

// ---- kernel: patch tokenizer (416 -> 128 -> 128) ---------------------------

__global__ __launch_bounds__(128)
void k_token(const _Float16* __restrict__ pfh, const int* __restrict__ cidx,
             const float* __restrict__ centers, const float* __restrict__ sf0,
             const float* __restrict__ sf1,
             const unsigned* __restrict__ Wf1, const float* __restrict__ b1,
             const float* __restrict__ g1, const float* __restrict__ be1,
             const unsigned* __restrict__ Wf2, const float* __restrict__ b2,
             const float* __restrict__ g2, const float* __restrict__ be2,
             float* __restrict__ out) {
  __shared__ alignas(16) _Float16 A[16 * 416];
  __shared__ alignas(16) float    Y[16 * TOK_];
  __shared__ alignas(16) _Float16 A2[16 * TOK_];
  int tid = threadIdx.x;
  int tile = blockIdx.x;               // 16 centers per block
  // async DMA of center features (dims 0..63) into A rows
  {
    int wv = __builtin_amdgcn_readfirstlane(tid >> 5);  // 4 waves
    int lane = tid & 31;
    if (lane < 16) {
      int bc = tile * 16 + lane;
      int b  = bc / NPATCH_;
      int ci = cidx[bc];
      unsigned long long sbase = (unsigned long long)(uintptr_t)pfh;
#pragma unroll
      for (int c = 0; c < 2; ++c) {
        int ch = wv + c * 4;             // 8 chunks of 16B per row
        unsigned ldsoff = (unsigned)(uintptr_t)&A[lane * 416 + ch * 8];
        unsigned voff =
            (unsigned)(((unsigned)(b * N_ + ci) * STEM_ + ch * 8) * 2u);
        async_chunk(ldsoff, voff, sbase);
      }
    }
  }
  // build dims 64..415 (scale feats + center PE) with VALU
  {
    int row = tid >> 3, seg = tid & 7; // 8 threads/row * 44 dims = 352
    int bc = tile * 16 + row;
    float ccx = centers[bc * 3 + 0], ccy = centers[bc * 3 + 1],
          ccz = centers[bc * 3 + 2];
    for (int d = 64 + seg * 44; d < 64 + seg * 44 + 44; ++d) {
      _Float16 v;
      if (d < 192) v = (_Float16)sf0[(size_t)bc * TOK_ + (d - 64)];
      else if (d < 320) v = (_Float16)sf1[(size_t)bc * TOK_ + (d - 192)];
      else {
        int q = d - 320, c = q >> 5, r = q & 31, i = r & 15;
        float fr = __expf(-(float)i * (9.2103403720f / 15.f));
        float coord = (c == 0) ? ccx : ((c == 1) ? ccy : ccz);
        float a = coord * fr;
        v = (_Float16)((r < 16) ? __sinf(a) : __cosf(a));
      }
      A[row * 416 + d] = v;
    }
  }
  async_wait0();
  __syncthreads();
  wmma_layer<416, 16, 128>(A, Wf1, Y, tid);
  __syncthreads();
  ln_rows_par<8, true, true>(Y, b1, g1, be1, A2, tid);
  __syncthreads();
  wmma_layer<128, 16, 128>(A2, Wf2, Y, tid);
  __syncthreads();
  ln_rows_par<8, false, false>(Y, b2, g2, be2, nullptr, tid);
  __syncthreads();
  if (tid < TOK_) {
    for (int rr = 0; rr < 16; ++rr)
      out[(size_t)(tile * 16 + rr) * TOK_ + tid] = Y[rr * TOK_ + tid];
  }
}

// ---- kernel: global token projection (B rows x 128 -> MLP2) ----------------

__device__ __forceinline__ float blk_sum128(float v, float* red, int tid) {
  red[tid] = v; __syncthreads();
  for (int s = 64; s > 0; s >>= 1) {
    if (tid < s) red[tid] += red[tid + s];
    __syncthreads();
  }
  float r = red[0]; __syncthreads();
  return r;
}

__global__ __launch_bounds__(128)
void k_proj(const unsigned* __restrict__ gmax,
            const float* __restrict__ W1, const float* __restrict__ b1,
            const float* __restrict__ g1, const float* __restrict__ be1,
            const float* __restrict__ W2, const float* __restrict__ b2,
            const float* __restrict__ g2, const float* __restrict__ be2,
            float* __restrict__ out) {
  __shared__ float x[TOK_];
  __shared__ float y[TOK_];
  __shared__ float red[TOK_];
  int tid = threadIdx.x, b = blockIdx.x;
  x[tid] = fdec(gmax[b * TOK_ + tid]);
  __syncthreads();
  float a = b1[tid];
  for (int i = 0; i < TOK_; ++i) a += x[i] * W1[i * TOK_ + tid];
  float s1 = blk_sum128(a, red, tid);
  float s2 = blk_sum128(a * a, red, tid);
  float mu = s1 / TOK_, var = s2 / TOK_ - mu * mu;
  float inv = rsqrtf(var + LN_EPS_);
  y[tid] = gelu_exact((a - mu) * inv * g1[tid] + be1[tid]);
  __syncthreads();
  float a2 = b2[tid];
  for (int i = 0; i < TOK_; ++i) a2 += y[i] * W2[i * TOK_ + tid];
  s1 = blk_sum128(a2, red, tid);
  s2 = blk_sum128(a2 * a2, red, tid);
  mu = s1 / TOK_; var = s2 / TOK_ - mu * mu; inv = rsqrtf(var + LN_EPS_);
  out[b * TOK_ + tid] = (a2 - mu) * inv * g2[tid] + be2[tid];
}

// ---------------------------------------------------------------------------

extern "C" void kernel_launch(void* const* d_in, const int* in_sizes, int n_in,
                              void* d_out, int out_size, void* d_ws,
                              size_t ws_size, hipStream_t stream) {
  // setup_inputs dict order: pointcloud, then params {stem1, stem2, scale0_1,
  // scale0_2, scale1_1, scale1_2, tok1, tok2, scene1, scene2, proj1, proj2},
  // each {W, b, g, be}.
  const float* pc = (const float*)d_in[0];
  const float* s1W  = (const float*)d_in[1];
  const float* s1b  = (const float*)d_in[2];
  const float* s1g  = (const float*)d_in[3];
  const float* s1be = (const float*)d_in[4];
  const float* s2W  = (const float*)d_in[5];
  const float* s2b  = (const float*)d_in[6];
  const float* s2g  = (const float*)d_in[7];
  const float* s2be = (const float*)d_in[8];
  const float* s01W = (const float*)d_in[9];
  const float* s01b = (const float*)d_in[10];
  const float* s01g = (const float*)d_in[11];
  const float* s01be = (const float*)d_in[12];
  const float* s02W = (const float*)d_in[13];
  const float* s02b = (const float*)d_in[14];
  const float* s02g = (const float*)d_in[15];
  const float* s02be = (const float*)d_in[16];
  const float* s11W = (const float*)d_in[17];
  const float* s11b = (const float*)d_in[18];
  const float* s11g = (const float*)d_in[19];
  const float* s11be = (const float*)d_in[20];
  const float* s12W = (const float*)d_in[21];
  const float* s12b = (const float*)d_in[22];
  const float* s12g = (const float*)d_in[23];
  const float* s12be = (const float*)d_in[24];
  const float* tk1W = (const float*)d_in[25];
  const float* tk1b = (const float*)d_in[26];
  const float* tk1g = (const float*)d_in[27];
  const float* tk1be = (const float*)d_in[28];
  const float* tk2W = (const float*)d_in[29];
  const float* tk2b = (const float*)d_in[30];
  const float* tk2g = (const float*)d_in[31];
  const float* tk2be = (const float*)d_in[32];
  const float* sc1W = (const float*)d_in[33];
  const float* sc1b = (const float*)d_in[34];
  const float* sc1g = (const float*)d_in[35];
  const float* sc1be = (const float*)d_in[36];
  const float* sc2W = (const float*)d_in[37];
  const float* sc2b = (const float*)d_in[38];
  const float* sc2g = (const float*)d_in[39];
  const float* sc2be = (const float*)d_in[40];
  const float* pj1W = (const float*)d_in[41];
  const float* pj1b = (const float*)d_in[42];
  const float* pj1g = (const float*)d_in[43];
  const float* pj1be = (const float*)d_in[44];
  const float* pj2W = (const float*)d_in[45];
  const float* pj2b = (const float*)d_in[46];
  const float* pj2g = (const float*)d_in[47];
  const float* pj2be = (const float*)d_in[48];

  // workspace carve (~2.8 MB)
  char* w = (char*)d_ws;
  _Float16* pfh     = (_Float16*)w; w += (size_t)B_ * N_ * STEM_ * 2;
  float*    centers = (float*)w;    w += (size_t)B_ * NPATCH_ * 3 * 4;
  int*      cidx    = (int*)w;      w += (size_t)B_ * NPATCH_ * 4;
  float*    sf0     = (float*)w;    w += (size_t)B_ * NPATCH_ * TOK_ * 4;
  float*    sf1     = (float*)w;    w += (size_t)B_ * NPATCH_ * TOK_ * 4;
  unsigned* gmax    = (unsigned*)w; w += (size_t)B_ * TOK_ * 4;
  unsigned* f_s01 = (unsigned*)w; w += (size_t)3 * 2048 * 4;
  unsigned* f_s02 = (unsigned*)w; w += (size_t)4 * 2048 * 4;
  unsigned* f_s11 = (unsigned*)w; w += (size_t)3 * 2048 * 4;
  unsigned* f_s12 = (unsigned*)w; w += (size_t)4 * 2048 * 4;
  unsigned* f_tk1 = (unsigned*)w; w += (size_t)13 * 2048 * 4;
  unsigned* f_tk2 = (unsigned*)w; w += (size_t)4 * 2048 * 4;
  unsigned* f_sc1 = (unsigned*)w; w += (size_t)6 * 2048 * 4;
  unsigned* f_sc2 = (unsigned*)w; w += (size_t)4 * 2048 * 4;

  float* out         = (float*)d_out;
  float* out_patch   = out;                 // B*64*128
  float* out_centers = out + 32768;         // B*64*3
  float* out_global  = out + 33536;         // B*1*128

  k_prep_wfrag<<<dim3(3, 8), 32, 0, stream>>>(s01W, 91, f_s01);
  k_prep_wfrag<<<dim3(4, 8), 32, 0, stream>>>(s02W, 128, f_s02);
  k_prep_wfrag<<<dim3(3, 8), 32, 0, stream>>>(s11W, 91, f_s11);
  k_prep_wfrag<<<dim3(4, 8), 32, 0, stream>>>(s12W, 128, f_s12);
  k_prep_wfrag<<<dim3(13, 8), 32, 0, stream>>>(tk1W, 416, f_tk1);
  k_prep_wfrag<<<dim3(4, 8), 32, 0, stream>>>(tk2W, 128, f_tk2);
  k_prep_wfrag<<<dim3(6, 8), 32, 0, stream>>>(sc1W, 187, f_sc1);
  k_prep_wfrag<<<dim3(4, 8), 32, 0, stream>>>(sc2W, 128, f_sc2);
  k_init_gmax<<<1, B_ * TOK_, 0, stream>>>(gmax);

  k_stem<<<B_ * N_ / 64, 64, 0, stream>>>(pc, s1W, s1b, s1g, s1be, s2W, s2b,
                                          s2g, s2be, pfh);
  k_fps<<<B_, 256, 0, stream>>>(pc, centers, cidx, out_centers);

  k_scale<16><<<B_ * NPATCH_, 128, 0, stream>>>(
      pc, pfh, centers, f_s01, s01b, s01g, s01be, f_s02, s02b, s02g, s02be,
      0.04f * 0.04f, sf0);
  k_scale<32><<<B_ * NPATCH_, 128, 0, stream>>>(
      pc, pfh, centers, f_s11, s11b, s11g, s11be, f_s12, s12b, s12g, s12be,
      0.08f * 0.08f, sf1);

  k_scene<<<B_ * N_, 256, 0, stream>>>(pc, pfh, f_sc1, sc1b, sc1g, sc1be,
                                       f_sc2, sc2b, sc2g, sc2be, gmax);

  k_token<<<B_ * NPATCH_ / 16, 128, 0, stream>>>(
      pfh, cidx, centers, sf0, sf1, f_tk1, tk1b, tk1g, tk1be, f_tk2, tk2b,
      tk2g, tk2be, out_patch);

  k_proj<<<B_, 128, 0, stream>>>(gmax, pj1W, pj1b, pj1g, pj1be, pj2W, pj2b,
                                 pj2g, pj2be, out_global);
  (void)in_sizes; (void)n_in; (void)out_size; (void)ws_size;
}
